// KokoroModel_49976239456694
// MI455X (gfx1250) — matrix operand order repaired and checked by reading, not compiled, not run on hardware
//
#include <hip/hip_runtime.h>
#include <hip/hip_bf16.h>
#include <math.h>

// =============================== types ======================================
typedef _Float16 f16;
typedef __attribute__((ext_vector_type(16))) _Float16 v16h;
typedef __attribute__((ext_vector_type(8)))  float    v8f;

union V16U { v16h h; unsigned u[8]; };

static __device__ __forceinline__ v8f wmma_f16(v16h a, v16h b, v8f c) {
  // D = A(16x32,f16) x B(32x16,f16) + C(16x16,f32)
  return __builtin_amdgcn_wmma_f32_16x16x32_f16(false, a, false, b, (short)0, c,
                                                false, false);
}

static __device__ __forceinline__ float sigmf(float x) {
  return 1.f / (1.f + __expf(-x));
}

// ===================== WMMA fragment loaders ================================
// A-matrix 16x32 f16 layout: lane = m + 16*g ; pair j -> k = 8g + (j%4)*2 + 16*(j/4)
static __device__ __forceinline__ v16h ldA_h(const f16* A, int lda, int m0,
                                             int k0, int lane) {
  int m = lane & 15, g = lane >> 4;
  const f16* row = A + (size_t)(m0 + m) * lda + k0 + 8 * g;
  V16U r;
  const unsigned* p0 = (const unsigned*)row;
  const unsigned* p1 = (const unsigned*)(row + 16);
  r.u[0] = p0[0]; r.u[1] = p0[1]; r.u[2] = p0[2]; r.u[3] = p0[3];
  r.u[4] = p1[0]; r.u[5] = p1[1]; r.u[6] = p1[2]; r.u[7] = p1[3];
  return r.h;
}

static __device__ __forceinline__ v16h ldA_h_g(const f16* A, int lda, int m0,
                                               int k0, int lane, int K) {
  int m = lane & 15, g = lane >> 4;
  const f16* row = A + (size_t)(m0 + m) * lda;
  v16h a;
#pragma unroll
  for (int j = 0; j < 8; ++j) {
    int k = k0 + 8 * g + (j & 3) * 2 + ((j >> 2) * 16);
    a[2 * j]     = (k     < K) ? row[k]     : (f16)0.f;
    a[2 * j + 1] = (k + 1 < K) ? row[k + 1] : (f16)0.f;
  }
  return a;
}

static __device__ __forceinline__ v16h ldA_f(const float* A, int lda, int m0,
                                             int k0, int lane) {
  int m = lane & 15, g = lane >> 4;
  const float* row = A + (size_t)(m0 + m) * lda;
  v16h a;
#pragma unroll
  for (int j = 0; j < 8; ++j) {
    int k = k0 + 8 * g + (j & 3) * 2 + ((j >> 2) * 16);
    a[2 * j]     = (f16)row[k];
    a[2 * j + 1] = (f16)row[k + 1];
  }
  return a;
}

static __device__ __forceinline__ v16h ldA_f_g(const float* A, int lda, int m0,
                                               int k0, int lane, int K) {
  int m = lane & 15, g = lane >> 4;
  const float* row = A + (size_t)(m0 + m) * lda;
  v16h a;
#pragma unroll
  for (int j = 0; j < 8; ++j) {
    int k = k0 + 8 * g + (j & 3) * 2 + ((j >> 2) * 16);
    a[2 * j]     = (k     < K) ? (f16)row[k]     : (f16)0.f;
    a[2 * j + 1] = (k + 1 < K) ? (f16)row[k + 1] : (f16)0.f;
  }
  return a;
}

// B-matrix 32x16 layout: lane = n + 16*g ; pair j -> k = 8g + (j%4)*2 + 16*(j/4)
// Source W is f16 row-major [N, K] (B[k][n] = W[n][k]); contiguous uint pairs.
static __device__ __forceinline__ v16h ldB_h(const f16* W, int ldw, int n0,
                                             int k0, int lane) {
  int n = lane & 15, g = lane >> 4;
  const unsigned* p =
      (const unsigned*)(W + (size_t)(n0 + n) * ldw + k0 + 8 * g);
  V16U r;
#pragma unroll
  for (int j = 0; j < 8; ++j) r.u[j] = p[(j & 3) + (j >> 2) * 8];
  return r.h;
}

static __device__ __forceinline__ v16h ldB_h_g(const f16* W, int ldw, int n0,
                                               int k0, int lane, int K) {
  int n = lane & 15, g = lane >> 4;
  const f16* row = W + (size_t)(n0 + n) * ldw;
  v16h b;
#pragma unroll
  for (int j = 0; j < 8; ++j) {
    int k = k0 + 8 * g + (j & 3) * 2 + ((j >> 2) * 16);
    b[2 * j]     = (k     < K) ? row[k]     : (f16)0.f;
    b[2 * j + 1] = (k + 1 < K) ? row[k + 1] : (f16)0.f;
  }
  return b;
}

// =============================== kernels ====================================
// GEMM: C[M,N] = act(A[M,K] @ W[N,K]^T + bias). Wave computes 16x64 (4 tiles,
// A-fragment reused across 4 WMMA). W is pre-converted f16.
template <bool A16, bool OUT16, bool RELU>
__global__ void k_gemm4(const void* Ap, int lda, const f16* W, int ldw,
                        const float* bias, void* Cp, int ldc, int M, int N,
                        int K) {
  int lane = threadIdx.x & 31, wv = threadIdx.x >> 5;
  int n0 = (blockIdx.x * 8 + wv) * 64, m0 = blockIdx.y * 16;
  if (n0 >= N) return;                       // wave-uniform
  int nt = (N - n0 + 15) >> 4; if (nt > 4) nt = 4;  // active 16-col tiles
  v8f acc[4] = {{}, {}, {}, {}};
  int kf = K & ~31;
  for (int k0 = 0; k0 < kf; k0 += 32) {
    v16h a;
    if constexpr (A16) a = ldA_h((const f16*)Ap, lda, m0, k0, lane);
    else               a = ldA_f((const float*)Ap, lda, m0, k0, lane);
#pragma unroll
    for (int i = 0; i < 4; ++i)
      if (i < nt) {
        v16h bb = ldB_h(W, ldw, n0 + 16 * i, k0, lane);
        acc[i] = wmma_f16(a, bb, acc[i]);
      }
  }
  if (kf < K) {
    v16h a;
    if constexpr (A16) a = ldA_h_g((const f16*)Ap, lda, m0, kf, lane, K);
    else               a = ldA_f_g((const float*)Ap, lda, m0, kf, lane, K);
#pragma unroll
    for (int i = 0; i < 4; ++i)
      if (i < nt) {
        v16h bb = ldB_h_g(W, ldw, n0 + 16 * i, kf, lane, K);
        acc[i] = wmma_f16(a, bb, acc[i]);
      }
  }
  int n = lane & 15, g = lane >> 4;
#pragma unroll
  for (int i = 0; i < 4; ++i) {
    if (i >= nt) break;
    int nc = n0 + 16 * i + n;
    float bv = bias ? bias[nc] : 0.f;
#pragma unroll
    for (int r = 0; r < 8; ++r) {
      float v = acc[i][r] + bv;
      if (RELU) v = fmaxf(v, 0.f);
      int m = r + 8 * g;
      if constexpr (OUT16)
        ((f16*)Cp)[(size_t)(m0 + m) * ldc + nc] = (f16)v;
      else
        ((float*)Cp)[(size_t)(m0 + m) * ldc + nc] = v;
    }
  }
}

// Persistent single-workgroup LSTM scan (B=16, H=512, 4H=2048).
__global__ __launch_bounds__(1024) void k_lstm(const float* xg, const f16* whh,
                                               const float* bhh, f16* out,
                                               int ldo, int T, int rev) {
  __shared__ f16 hls[16 * 512];
  __shared__ float cls[16 * 512];
  int tid = threadIdx.x, lane = tid & 31, wv = tid >> 5;  // 32 waves
  for (int i = tid; i < 16 * 512; i += 1024) {
    hls[i] = (f16)0.f;
    cls[i] = 0.f;
  }
  __syncthreads();
  int n = lane & 15, g = lane >> 4;
  int col = wv * 16 + n;  // wave owns cols [wv*16, wv*16+16) in all 4 gates
  for (int s = 0; s < T; ++s) {
    int t = rev ? (T - 1 - s) : s;
    v8f a0 = {}, a1 = {}, a2 = {}, a3 = {};
    for (int k0 = 0; k0 < 512; k0 += 32) {
      v16h av = ldA_h(hls, 512, 0, k0, lane);
      v16h b0 = ldB_h(whh, 512, 0 * 512 + wv * 16, k0, lane);
      a0 = wmma_f16(av, b0, a0);
      v16h b1 = ldB_h(whh, 512, 1 * 512 + wv * 16, k0, lane);
      a1 = wmma_f16(av, b1, a1);
      v16h b2 = ldB_h(whh, 512, 2 * 512 + wv * 16, k0, lane);
      a2 = wmma_f16(av, b2, a2);
      v16h b3 = ldB_h(whh, 512, 3 * 512 + wv * 16, k0, lane);
      a3 = wmma_f16(av, b3, a3);
    }
    __syncthreads();  // all h reads done before updates
#pragma unroll
    for (int r = 0; r < 8; ++r) {
      int m = r + 8 * g;  // batch row
      size_t xrow = ((size_t)m * T + t) * 2048;
      float iv = a0[r] + xg[xrow + col]        + bhh[col];
      float fv = a1[r] + xg[xrow + 512 + col]  + bhh[512 + col];
      float gv = a2[r] + xg[xrow + 1024 + col] + bhh[1024 + col];
      float ov = a3[r] + xg[xrow + 1536 + col] + bhh[1536 + col];
      float c = cls[m * 512 + col];
      c = sigmf(fv) * c + sigmf(iv) * tanhf(gv);
      float hh = sigmf(ov) * tanhf(c);
      cls[m * 512 + col] = c;
      hls[m * 512 + col] = (f16)hh;
      out[((size_t)m * T + t) * (size_t)ldo + col] = (f16)hh;
    }
    __syncthreads();
  }
}

// Fused attention: one block per (b, h, 16-query tile). S row kept in LDS.
// Vt is V pre-transposed to [b,h,d,t] so both phases use contiguous B loads.
__global__ __launch_bounds__(128) void k_attn(const f16* Q, const f16* Kb,
                                              const f16* Vt, const float* maskb,
                                              f16* att) {
  __shared__ float S[16][512];
  __shared__ f16 P[16][512];
  __shared__ float red[16][8];
  __shared__ float stat[16];
  int b = blockIdx.z, h = blockIdx.y, tq0 = blockIdx.x * 16;
  int tid = threadIdx.x, lane = tid & 31, w = tid >> 5;
  const float scale = 0.125f;  // 1/sqrt(64)
  v16h qa0 = ldA_h(Q + h * 64, 512, b * 512 + tq0, 0, lane);
  v16h qa1 = ldA_h(Q + h * 64, 512, b * 512 + tq0, 32, lane);
#pragma unroll
  for (int j2 = 0; j2 < 8; ++j2) {
    int tk0 = w * 128 + j2 * 16;
    v16h b0 = ldB_h(Kb + h * 64, 512, b * 512 + tk0, 0, lane);
    v16h b1 = ldB_h(Kb + h * 64, 512, b * 512 + tk0, 32, lane);
    v8f acc = {};
    acc = wmma_f16(qa0, b0, acc);
    acc = wmma_f16(qa1, b1, acc);
    int n = lane & 15, g = lane >> 4;
    float bv = maskb[b * 512 + tk0 + n];
#pragma unroll
    for (int r = 0; r < 8; ++r) S[r + 8 * g][tk0 + n] = acc[r] * scale + bv;
  }
  __syncthreads();
  // row softmax (8 threads per row)
  int row = tid >> 3, l8 = tid & 7;
  float mx = -3.4e38f;
  for (int c = l8 * 64; c < l8 * 64 + 64; ++c) mx = fmaxf(mx, S[row][c]);
  red[row][l8] = mx;
  __syncthreads();
  if (l8 == 0) {
    float m2 = red[row][0];
    for (int i = 1; i < 8; ++i) m2 = fmaxf(m2, red[row][i]);
    stat[row] = m2;
  }
  __syncthreads();
  float rm = stat[row], sum = 0.f;
  for (int c = l8 * 64; c < l8 * 64 + 64; ++c) {
    float e = __expf(S[row][c] - rm);
    S[row][c] = e;
    sum += e;
  }
  red[row][l8] = sum;
  __syncthreads();
  if (l8 == 0) {
    float s2 = 0.f;
    for (int i = 0; i < 8; ++i) s2 += red[row][i];
    stat[row] = s2;
  }
  __syncthreads();
  float inv = 1.f / stat[row];
  for (int c = l8 * 64; c < l8 * 64 + 64; ++c)
    P[row][c] = (f16)(S[row][c] * inv);
  __syncthreads();
  // att[16, 64] = P[16,512] @ V[512,64] ; wave w owns d-range [w*16, w*16+16)
  int d0 = w * 16;
  const f16* Vrow = Vt + ((size_t)(b * 8 + h) * 64) * 512;  // [d][t] rows
  v8f acc = {};
  for (int k0 = 0; k0 < 512; k0 += 32) {
    v16h a = ldA_h(&P[0][0], 512, 0, k0, lane);
    v16h bb = ldB_h(Vrow, 512, d0, k0, lane);  // B[k=t][n=d] = Vt[d][t]
    acc = wmma_f16(a, bb, acc);
  }
  int n = lane & 15, g = lane >> 4;
#pragma unroll
  for (int r = 0; r < 8; ++r) {
    int m = r + 8 * g;
    att[((size_t)(b * 512 + tq0 + m)) * 512 + h * 64 + d0 + n] = (f16)acc[r];
  }
}

__global__ void k_embed(const int* idx, const float* emb, f16* X) {
  int i = blockIdx.x;  // row in [0, 2048)
  int v = idx[i];
  for (int j = threadIdx.x; j < 512; j += 256)
    X[(size_t)i * 512 + j] = (f16)emb[(size_t)v * 512 + j];
}

__global__ void k_cvt(const float* src, f16* dst, int n) {
  int i = blockIdx.x * 256 + threadIdx.x;
  if (i < n) dst[i] = (f16)src[i];
}

// V[b*512+t][h*64+d] -> Vt[((b*8+h)*64+d)*512 + t]
__global__ void k_vtr(const f16* Vb, f16* Vt) {
  int i = blockIdx.x * 256 + threadIdx.x;
  if (i >= 16 * 512 * 512) return;
  int d = i & 511, r = i >> 9;
  int t = r & 511, b = r >> 9;
  int h = d >> 6, dd = d & 63;
  Vt[(((size_t)(b * 8 + h)) * 64 + dd) * 512 + t] = Vb[(size_t)r * 512 + d];
}

__global__ void k_shift(const float* mel, f16* sh) {
  int i = blockIdx.x * 256 + threadIdx.x;
  if (i >= 16 * 512 * 80) return;
  int c = i % 80, r = i / 80;
  int t = r % 512, b = r / 512;
  sh[i] = (t == 0) ? (f16)0.f : (f16)mel[((size_t)b * 512 + t - 1) * 80 + c];
}

// Length regulation: round/cumsum/searchsorted + masked gather.
__global__ __launch_bounds__(512) void k_lenreg(const float* dur,
                                                const float* enc, f16* expb,
                                                float* maskb) {
  __shared__ int cum[128];
  __shared__ int idxs[512];
  int b = blockIdx.x, tid = threadIdx.x;
  if (tid < 128) cum[tid] = (int)rintf(dur[b * 128 + tid]);  // round-half-even
  __syncthreads();
  if (tid == 0) {
    int s = 0;
    for (int i = 0; i < 128; ++i) { s += cum[i]; cum[i] = s; }
  }
  __syncthreads();
  int pos = tid, total = cum[127];
  int lo = 0, hi = 128;
  while (lo < hi) {  // searchsorted side='right'
    int mid = (lo + hi) >> 1;
    if (cum[mid] <= pos) lo = mid + 1; else hi = mid;
  }
  int idx = lo > 127 ? 127 : lo;
  bool masked = pos >= total;
  idxs[pos] = masked ? -1 : idx;
  maskb[b * 512 + pos] = masked ? -1.0e9f : 0.f;
  __syncthreads();
  for (int p = 0; p < 512; ++p) {
    int id = idxs[p];
    const float* src = enc + ((size_t)b * 128 + (id < 0 ? 0 : id)) * 512;
    f16* dst = expb + ((size_t)b * 512 + p) * 512;
    dst[tid] = (id < 0) ? (f16)0.f : (f16)src[tid];
  }
}

// out[row] = dot(A[row, :K], w) + bias[0]
__global__ __launch_bounds__(128) void k_rowdot(const f16* A, int lda,
                                                const float* w, int K,
                                                const float* bias, float* out) {
  __shared__ float sred[128];
  int row = blockIdx.x;
  float s = 0.f;
  for (int k = threadIdx.x; k < K; k += 128)
    s += (float)A[(size_t)row * lda + k] * w[k];
  sred[threadIdx.x] = s;
  __syncthreads();
  for (int off = 64; off > 0; off >>= 1) {
    if (threadIdx.x < off) sred[threadIdx.x] += sred[threadIdx.x + off];
    __syncthreads();
  }
  if (threadIdx.x == 0) out[row] = sred[0] + bias[0];
}

// =============================== host =======================================
extern "C" void kernel_launch(void* const* d_in, const int* in_sizes, int n_in,
                              void* d_out, int out_size, void* d_ws,
                              size_t ws_size, hipStream_t stream) {
  const int* phon      = (const int*)d_in[0];
  const float* mel     = (const float*)d_in[1];
  const float* dur     = (const float*)d_in[2];
  // d_in[3] stop_token_targets: unused
  const float* emb     = (const float*)d_in[4];
  const float* ef_wih  = (const float*)d_in[5];
  const float* ef_whh  = (const float*)d_in[6];
  const float* ef_bih  = (const float*)d_in[7];
  const float* ef_bhh  = (const float*)d_in[8];
  const float* eb_wih  = (const float*)d_in[9];
  const float* eb_whh  = (const float*)d_in[10];
  const float* eb_bih  = (const float*)d_in[11];
  const float* eb_bhh  = (const float*)d_in[12];
  const float* proj_w  = (const float*)d_in[13];
  const float* proj_b  = (const float*)d_in[14];
  const float* dur_w1  = (const float*)d_in[15];
  const float* dur_b1  = (const float*)d_in[16];
  const float* dur_w2  = (const float*)d_in[17];
  const float* dur_b2  = (const float*)d_in[18];
  const float* dur_w3  = (const float*)d_in[19];
  const float* dur_b3  = (const float*)d_in[20];
  const float* min_w   = (const float*)d_in[21];
  const float* min_b   = (const float*)d_in[22];
  const float* dec_wih = (const float*)d_in[23];
  const float* dec_whh = (const float*)d_in[24];
  const float* dec_bih = (const float*)d_in[25];
  const float* dec_bhh = (const float*)d_in[26];
  const float* ain_w   = (const float*)d_in[27];
  const float* ain_b   = (const float*)d_in[28];
  const float* aout_w  = (const float*)d_in[29];
  const float* aout_b  = (const float*)d_in[30];
  const float* mout_w  = (const float*)d_in[31];
  const float* mout_b  = (const float*)d_in[32];
  const float* stop_w  = (const float*)d_in[33];
  const float* stop_b  = (const float*)d_in[34];

  float* out = (float*)d_out;  // mels[655360] | log_dur[2048] | stops[8192]

  const size_t MBv = 1ull << 20;
  char* w = (char*)d_ws;
  // Activations (xgdec reuses the 2..66MB region whose tenants are dead by then)
  f16*   X      = (f16*)(w + 0 * MBv);        // [2048,512]
  float* enc    = (float*)(w + 2 * MBv);      // [2048,512]
  float* maskb  = (float*)(w + 6 * MBv);      // [16,512]
  f16*   encin  = (f16*)(w + 7 * MBv);        // [2048,1024]
  float* xgf    = (float*)(w + 11 * MBv);     // [2048,2048]
  float* xgb    = (float*)(w + 27 * MBv);     // [2048,2048]
  f16*   h1     = (f16*)(w + 43 * MBv);       // [2048,512]
  f16*   h2     = (f16*)(w + 45 * MBv);       // [2048,256]
  f16*   expb   = (f16*)(w + 46 * MBv);       // [8192,512]
  f16*   Kb     = (f16*)(w + 54 * MBv);       // [8192,512]
  f16*   Vb     = (f16*)(w + 62 * MBv);       // [8192,512]
  f16*   Qb     = (f16*)(w + 70 * MBv);       // [8192,512]
  f16*   shifted= (f16*)(w + 78 * MBv);       // [8192,80]
  f16*   xall   = (f16*)(w + 80 * MBv);       // [8192,1024] = [melp | att_out]
  f16*   attb   = (f16*)(w + 96 * MBv);       // [8192,512]
  f16*   Vt     = (f16*)(w + 104 * MBv);      // [16,8,64,512]
  float* xgdec  = (float*)(w + 2 * MBv);      // [8192,2048] overlap
  // Pre-converted f16 weights
  f16* wFwhh = (f16*)(w + 112 * MBv);         // [2048,512]
  f16* wBwhh = (f16*)(w + 114 * MBv);
  f16* wDwhh = (f16*)(w + 116 * MBv);
  f16* wFwih = (f16*)(w + 118 * MBv);         // [2048,512]
  f16* wBwih = (f16*)(w + 120 * MBv);
  f16* wProj = (f16*)(w + 122 * MBv);         // [512,1024]
  f16* wDur1 = (f16*)(w + 123 * MBv);         // [512,512]  (0.5MB)
  f16* wDur2 = (f16*)(w + 123 * MBv + 512 * 1024);  // [256,512] (0.25MB)
  f16* wMin  = (f16*)(w + 124 * MBv);         // [512,80]
  f16* wDecWih = (f16*)(w + 125 * MBv);       // [2048,1024] (4MB)
  f16* wAin  = (f16*)(w + 129 * MBv);         // [1536,512] (1.5MB)
  f16* wAout = (f16*)(w + 131 * MBv);         // [512,512]
  f16* wMout = (f16*)(w + 132 * MBv);         // [80,512]
  f16* hall  = (f16*)(w + 133 * MBv);         // [8192,512] -> ends 141MB

  // --- Weight conversion (once) ---
  k_cvt<<<(2048*512+255)/256, 256, 0, stream>>>(ef_whh, wFwhh, 2048*512);
  k_cvt<<<(2048*512+255)/256, 256, 0, stream>>>(eb_whh, wBwhh, 2048*512);
  k_cvt<<<(2048*512+255)/256, 256, 0, stream>>>(dec_whh, wDwhh, 2048*512);
  k_cvt<<<(2048*512+255)/256, 256, 0, stream>>>(ef_wih, wFwih, 2048*512);
  k_cvt<<<(2048*512+255)/256, 256, 0, stream>>>(eb_wih, wBwih, 2048*512);
  k_cvt<<<(512*1024+255)/256, 256, 0, stream>>>(proj_w, wProj, 512*1024);
  k_cvt<<<(512*512+255)/256, 256, 0, stream>>>(dur_w1, wDur1, 512*512);
  k_cvt<<<(256*512+255)/256, 256, 0, stream>>>(dur_w2, wDur2, 256*512);
  k_cvt<<<(512*80+255)/256, 256, 0, stream>>>(min_w, wMin, 512*80);
  k_cvt<<<(2048*1024+255)/256, 256, 0, stream>>>(dec_wih, wDecWih, 2048*1024);
  k_cvt<<<(1536*512+255)/256, 256, 0, stream>>>(ain_w, wAin, 1536*512);
  k_cvt<<<(512*512+255)/256, 256, 0, stream>>>(aout_w, wAout, 512*512);
  k_cvt<<<(80*512+255)/256, 256, 0, stream>>>(mout_w, wMout, 80*512);

  // --- Encoder ---
  k_embed<<<2048, 256, 0, stream>>>(phon, emb, X);
  k_gemm4<true, false, false><<<dim3(4, 128), 256, 0, stream>>>(
      X, 512, wFwih, 512, ef_bih, xgf, 2048, 2048, 2048, 512);
  k_gemm4<true, false, false><<<dim3(4, 128), 256, 0, stream>>>(
      X, 512, wBwih, 512, eb_bih, xgb, 2048, 2048, 2048, 512);
  k_lstm<<<1, 1024, 0, stream>>>(xgf, wFwhh, ef_bhh, encin, 1024, 128, 0);
  k_lstm<<<1, 1024, 0, stream>>>(xgb, wBwhh, eb_bhh, encin + 512, 1024, 128, 1);
  k_gemm4<true, false, false><<<dim3(1, 128), 256, 0, stream>>>(
      encin, 1024, wProj, 1024, proj_b, enc, 512, 2048, 512, 1024);

  // --- Duration predictor ---
  k_gemm4<false, true, true><<<dim3(1, 128), 256, 0, stream>>>(
      enc, 512, wDur1, 512, dur_b1, h1, 512, 2048, 512, 512);
  k_gemm4<true, true, true><<<dim3(1, 128), 256, 0, stream>>>(
      h1, 512, wDur2, 512, dur_b2, h2, 256, 2048, 256, 512);
  k_rowdot<<<2048, 128, 0, stream>>>(h2, 256, dur_w3, 256, dur_b3,
                                     out + 655360);

  // --- Length regulate + attention inputs ---
  k_lenreg<<<16, 512, 0, stream>>>(dur, enc, expb, maskb);
  k_gemm4<true, true, false><<<dim3(1, 512), 256, 0, stream>>>(
      expb, 512, wAin + 512 * 512, 512, ain_b + 512, Kb, 512, 8192, 512, 512);
  k_gemm4<true, true, false><<<dim3(1, 512), 256, 0, stream>>>(
      expb, 512, wAin + 1024 * 512, 512, ain_b + 1024, Vb, 512, 8192, 512, 512);
  k_vtr<<<(16*512*512 + 255) / 256, 256, 0, stream>>>(Vb, Vt);
  k_shift<<<(16*512*80 + 255) / 256, 256, 0, stream>>>(mel, shifted);
  k_gemm4<true, true, false><<<dim3(1, 512), 256, 0, stream>>>(
      shifted, 80, wMin, 80, min_b, xall, 1024, 8192, 512, 80);  // melp
  k_gemm4<true, true, false><<<dim3(1, 512), 256, 0, stream>>>(
      xall, 1024, wAin, 512, ain_b, Qb, 512, 8192, 512, 512);    // Q

  // --- Attention (hoisted out of the decoder scan) ---
  k_attn<<<dim3(32, 8, 16), 128, 0, stream>>>(Qb, Kb, Vt, maskb, attb);
  k_gemm4<true, true, false><<<dim3(1, 512), 256, 0, stream>>>(
      attb, 512, wAout, 512, aout_b, xall + 512, 1024, 8192, 512, 512);

  // --- Decoder: precomputed input gates + sequential LSTM scan ---
  k_gemm4<true, false, false><<<dim3(4, 512), 256, 0, stream>>>(
      xall, 1024, wDecWih, 1024, dec_bih, xgdec, 2048, 8192, 2048, 1024);
  k_lstm<<<1, 1024, 0, stream>>>(xgdec, wDwhh, dec_bhh, hall, 512, 512, 0);

  // --- Output heads ---
  k_gemm4<true, false, false><<<dim3(1, 512), 256, 0, stream>>>(
      hall, 512, wMout, 512, mout_b, out, 80, 8192, 80, 512);
  k_rowdot<<<8192, 128, 0, stream>>>(hall, 512, stop_w, 512, stop_b,
                                     out + 657408);
}
